// CellRetrievalNetwork_71408126263863
// MI455X (gfx1250) — matrix operand order repaired and checked by reading, not compiled, not run on hardware
//
#include <hip/hip_runtime.h>

#define B_ 8
#define N_ 2048
#define D_ 256
#define K_ 8
#define P_ (B_ * N_)   // 16384 points
#define E_ (P_ * K_)   // 131072 edges

typedef __attribute__((ext_vector_type(16))) _Float16 v16h;
typedef __attribute__((ext_vector_type(8)))  _Float16 v8h;
typedef __attribute__((ext_vector_type(8)))  float    v8f;

// ---------------------------------------------------------------- WMMA utils

static __device__ __forceinline__ v8f wmma_f16(v16h a, v16h b, v8f c) {
  // D = A(16x32 f16) * B(32x16 f16) + C(16x16 f32)
  return __builtin_amdgcn_wmma_f32_16x16x32_f16(
      /*neg_a=*/false, a, /*neg_b=*/false, b,
      /*c_mod=*/(short)0, c, /*reuse_a=*/false, /*reuse_b=*/false);
}

static __device__ __forceinline__ v16h make_a(v8h lo, v8h hi) {
  v16h a;
#pragma unroll
  for (int t = 0; t < 8; ++t) { a[t] = lo[t]; a[t + 8] = hi[t]; }
  return a;
}

// A-fragment: lane row M=lane%16; laneHi(=lane>>4) selects K sub-blocks
// {0..7,16..23} (laneHi=0) or {8..15,24..31} (laneHi=1) per ISA layout.
static __device__ __forceinline__ v16h load_a(const _Float16* row, int kc, int lh) {
  v8h lo = *(const v8h*)(row + kc + lh * 8);
  v8h hi = *(const v8h*)(row + kc + 16 + lh * 8);
  return make_a(lo, hi);
}

// B-fragment: lane holds column N=lane%16 (a row of the transposed operand);
// laneHi selects K sub-blocks {0..7,16..23} or {8..15,24..31}, same as A.
static __device__ __forceinline__ v16h load_b(const _Float16* colrow, int kc, int lh) {
  v8h lo = *(const v8h*)(colrow + kc + lh * 8);
  v8h hi = *(const v8h*)(colrow + kc + 16 + lh * 8);
  return make_a(lo, hi);
}

// 8 halves of the virtual edge row e=[xi | xj-xi] at half-offset `off` (0..511)
static __device__ __forceinline__ v8h load_e8(const _Float16* xi, const _Float16* xj, int off) {
  if (off < 256) return *(const v8h*)(xi + off);
  v8h a = *(const v8h*)(xi + off - 256);
  v8h b = *(const v8h*)(xj + off - 256);
  v8h r;
#pragma unroll
  for (int t = 0; t < 8; ++t) r[t] = (_Float16)((float)b[t] - (float)a[t]);
  return r;
}

// 8 halves of BN-normalized h1 at half-offset `off` (per-channel affine)
static __device__ __forceinline__ v8h load_bn8(const _Float16* h, const float* sc,
                                               const float* sh, int off) {
  v8h r = *(const v8h*)(h + off);
#pragma unroll
  for (int t = 0; t < 8; ++t) r[t] = (_Float16)((float)r[t] * sc[off + t] + sh[off + t]);
  return r;
}

static __device__ __forceinline__ void atomicMaxFloat(float* addr, float val) {
  // IEEE trick: positive floats order as ints, negatives reversed as uints.
  if (val >= 0.0f) atomicMax((int*)addr, __float_as_int(val));
  else             atomicMin((unsigned int*)addr, __float_as_uint(val));
}

// ---------------------------------------------------------------- prep

__global__ __launch_bounds__(256) void transpose_f16_kernel(
    const float* __restrict__ W, _Float16* __restrict__ Wt, int Krows, int Ncols) {
  int t = blockIdx.x * blockDim.x + threadIdx.x;
  if (t >= Krows * Ncols) return;
  int n = t / Krows, k = t % Krows;
  Wt[t] = (_Float16)W[(size_t)k * Ncols + n];  // Wt[n][k] = W[k][n]
}

__global__ __launch_bounds__(256) void init_kernel(
    float* __restrict__ pooled, float* __restrict__ s1, float* __restrict__ q1,
    float* __restrict__ s2, float* __restrict__ q2) {
  int t = blockIdx.x * blockDim.x + threadIdx.x;
  if (t < B_ * D_) pooled[t] = __uint_as_float(0xFF800000u);  // -inf
  if (t < D_) { s1[t] = 0.f; q1[t] = 0.f; s2[t] = 0.f; q2[t] = 0.f; }
}

// ------------------------------------------------- per-point embeddings

static __device__ __forceinline__ void mlp_branch(
    const float* __restrict__ in3, const float* __restrict__ W1, const float* __restrict__ b1,
    const float* __restrict__ W2, const float* __restrict__ b2, int lane,
    _Float16* __restrict__ dst) {
  float i0 = in3[0], i1 = in3[1], i2 = in3[2];
  float hid = fmaxf(i0 * W1[lane] + i1 * W1[32 + lane] + i2 * W1[64 + lane] + b1[lane], 0.f);
  float acc[8];
#pragma unroll
  for (int t = 0; t < 8; ++t) acc[t] = b2[lane + 32 * t];
  for (int j = 0; j < 32; ++j) {
    float hj = __shfl(hid, j, 32);
    const float* w = W2 + j * 256;
#pragma unroll
    for (int t = 0; t < 8; ++t) acc[t] += hj * w[lane + 32 * t];
  }
  float ss = 0.f;
#pragma unroll
  for (int t = 0; t < 8; ++t) { acc[t] = fmaxf(acc[t], 0.f); ss += acc[t] * acc[t]; }
#pragma unroll
  for (int m = 16; m; m >>= 1) ss += __shfl_xor(ss, m, 32);
  float inv = 1.f / fmaxf(sqrtf(ss), 1e-12f);
#pragma unroll
  for (int t = 0; t < 8; ++t) dst[lane + 32 * t] = (_Float16)(acc[t] * inv);
}

__global__ __launch_bounds__(256) void embed_kernel(
    const int* __restrict__ cls, const float* __restrict__ colors,
    const float* __restrict__ positions, const float* __restrict__ table,
    const float* __restrict__ posW1, const float* __restrict__ posb1,
    const float* __restrict__ posW2, const float* __restrict__ posb2,
    const float* __restrict__ colW1, const float* __restrict__ colb1,
    const float* __restrict__ colW2, const float* __restrict__ colb2,
    _Float16* __restrict__ feat) {
  int gw = (blockIdx.x * blockDim.x + threadIdx.x) >> 5;  // one wave per point
  int lane = threadIdx.x & 31;
  if (gw >= P_) return;
  _Float16* frow = feat + (size_t)gw * 768;

  {  // class embedding + l2norm -> feat[0:256)
    int ci = cls[gw];
    const float* trow = table + ci * D_;
    float v[8]; float ss = 0.f;
#pragma unroll
    for (int t = 0; t < 8; ++t) { float x = trow[lane + 32 * t]; v[t] = x; ss += x * x; }
#pragma unroll
    for (int m = 16; m; m >>= 1) ss += __shfl_xor(ss, m, 32);
    float inv = 1.f / fmaxf(sqrtf(ss), 1e-12f);
#pragma unroll
    for (int t = 0; t < 8; ++t) frow[lane + 32 * t] = (_Float16)(v[t] * inv);
  }
  mlp_branch(colors + gw * 3, colW1, colb1, colW2, colb2, lane, frow + 256);
  mlp_branch(positions + gw * 3, posW1, posb1, posW2, posb2, lane, frow + 512);
}

// ------------------------------------------------- merge GEMM 16384x768x256
// 16x64 tile per wave, software-pipelined K loop: fragments for chunk kc+32
// are in flight while the 4 WMMAs for chunk kc execute.

__global__ __launch_bounds__(128) void merge_gemm_kernel(
    const _Float16* __restrict__ feat, const _Float16* __restrict__ Wt,
    const float* __restrict__ bias, _Float16* __restrict__ xh) {
  int lane = threadIdx.x, lh = lane >> 4, ll = lane & 15;
  int rt = blockIdx.x * 4 + threadIdx.y;  // row tile
  int ctg = blockIdx.y;                   // 64-column group
  const _Float16* arow = feat + (size_t)(rt * 16 + ll) * 768;
  const _Float16* brow = Wt + (size_t)(ctg * 64 + ll) * 768;
  v8f c[4] = {{}, {}, {}, {}};
  v16h a_cur = load_a(arow, 0, lh);
  v16h b_cur[4];
#pragma unroll
  for (int u = 0; u < 4; ++u) b_cur[u] = load_b(brow + (size_t)u * 16 * 768, 0, lh);
#pragma unroll
  for (int kc = 0; kc < 768; kc += 32) {
    v16h a_nxt = a_cur;
    v16h b_nxt[4] = {b_cur[0], b_cur[1], b_cur[2], b_cur[3]};
    if (kc + 32 < 768) {
      a_nxt = load_a(arow, kc + 32, lh);
#pragma unroll
      for (int u = 0; u < 4; ++u)
        b_nxt[u] = load_b(brow + (size_t)u * 16 * 768, kc + 32, lh);
    }
#pragma unroll
    for (int u = 0; u < 4; ++u) c[u] = wmma_f16(a_cur, b_cur[u], c[u]);
    a_cur = a_nxt;
#pragma unroll
    for (int u = 0; u < 4; ++u) b_cur[u] = b_nxt[u];
  }
  int rbase = rt * 16 + 8 * lh;
#pragma unroll
  for (int u = 0; u < 4; ++u) {
    int col = ctg * 64 + u * 16 + ll;
    float bv = bias[col];
#pragma unroll
    for (int v = 0; v < 8; ++v)
      xh[(size_t)(rbase + v) * 256 + col] = (_Float16)fmaxf(c[u][v] + bv, 0.f);
  }
}

__global__ __launch_bounds__(256) void sq_kernel(const _Float16* __restrict__ xh,
                                                 float* __restrict__ sq) {
  int gw = (blockIdx.x * blockDim.x + threadIdx.x) >> 5;
  int lane = threadIdx.x & 31;
  if (gw >= P_) return;
  const _Float16* row = xh + (size_t)gw * 256;
  float ss = 0.f;
#pragma unroll
  for (int t = 0; t < 8; ++t) { float v = (float)row[lane + 32 * t]; ss += v * v; }
#pragma unroll
  for (int m = 16; m; m >>= 1) ss += __shfl_xor(ss, m, 32);
  if (lane == 0) sq[gw] = ss;
}

// ------------------------------------------------- fused Gram + top-8 kNN

__global__ __launch_bounds__(32) void knn_kernel(const _Float16* __restrict__ xh,
                                                 const float* __restrict__ sq,
                                                 int* __restrict__ knn) {
  __shared__ float S[16][17];
  int lane = threadIdx.x, lh = lane >> 4, ll = lane & 15;
  const int tiles = N_ / 16;
  int it = blockIdx.x % tiles;
  int b  = blockIdx.x / tiles;
  int base = b * N_;
  const _Float16* arow = xh + (size_t)(base + it * 16 + ll) * 256;
  v16h afrag[8];
#pragma unroll
  for (int kc = 0; kc < 8; ++kc) afrag[kc] = load_a(arow, kc * 32, lh);

  float best[8]; int bidx[8];
#pragma unroll
  for (int m = 0; m < 8; ++m) { best[m] = 3.0e38f; bidx[m] = 0; }
  float sqi = (lane < 16) ? sq[base + it * 16 + lane] : 0.f;

  for (int jt = 0; jt < tiles; ++jt) {
    const _Float16* brow = xh + (size_t)(base + jt * 16 + ll) * 256;
    v8f c = {};
#pragma unroll
    for (int kc = 0; kc < 8; ++kc)
      c = wmma_f16(afrag[kc], load_b(brow, kc * 32, lh), c);
#pragma unroll
    for (int v = 0; v < 8; ++v) S[v + 8 * lh][ll] = c[v];  // S[i_local][j_local]
    __syncthreads();
    if (lane < 16) {
#pragma unroll
      for (int n = 0; n < 16; ++n) {
        int j = jt * 16 + n;
        float d = sqi + sq[base + j] - 2.f * S[lane][n];
        if (d < best[7]) {
          int pos = 7;
#pragma unroll
          for (int m = 6; m >= 0; --m) if (d < best[m]) pos = m;
          for (int m = 7; m > pos; --m) { best[m] = best[m - 1]; bidx[m] = bidx[m - 1]; }
          best[pos] = d; bidx[pos] = j;
        }
      }
    }
    __syncthreads();
  }
  if (lane < 16) {
    int* o = knn + (size_t)(base + it * 16 + lane) * K_;
#pragma unroll
    for (int m = 0; m < 8; ++m) o[m] = bidx[m];
  }
}

// -------------- edge GEMM 1: 131072 x 512 x 256 (fused gather, 16x64 tiles,
// software-pipelined: next chunk's gather/subtract VALU work + loads overlap
// the current chunk's 4 WMMAs).

__global__ __launch_bounds__(128) void edge_gemm1_kernel(
    const _Float16* __restrict__ xh, const int* __restrict__ knn,
    const _Float16* __restrict__ W1t, const float* __restrict__ b1,
    _Float16* __restrict__ h1, float* __restrict__ sum1, float* __restrict__ sumsq1) {
  int lane = threadIdx.x, lh = lane >> 4, ll = lane & 15;
  int rt = blockIdx.x * 4 + threadIdx.y;
  int ctg = blockIdx.y;
  int er = rt * 16 + ll;        // edge row
  int node = er >> 3;           // /K
  int k = er & 7;
  int b = node >> 11;           // /N
  int j = knn[node * K_ + k];   // local j in [0,N)
  const _Float16* xi = xh + (size_t)node * 256;
  const _Float16* xj = xh + (size_t)(b * N_ + j) * 256;
  const _Float16* brow = W1t + (size_t)(ctg * 64 + ll) * 512;
  v8f c[4] = {{}, {}, {}, {}};
  v16h a_cur = make_a(load_e8(xi, xj, lh * 8), load_e8(xi, xj, 16 + lh * 8));
  v16h b_cur[4];
#pragma unroll
  for (int u = 0; u < 4; ++u) b_cur[u] = load_b(brow + (size_t)u * 16 * 512, 0, lh);
#pragma unroll
  for (int kc = 0; kc < 512; kc += 32) {
    v16h a_nxt = a_cur;
    v16h b_nxt[4] = {b_cur[0], b_cur[1], b_cur[2], b_cur[3]};
    if (kc + 32 < 512) {
      a_nxt = make_a(load_e8(xi, xj, kc + 32 + lh * 8),
                     load_e8(xi, xj, kc + 48 + lh * 8));
#pragma unroll
      for (int u = 0; u < 4; ++u)
        b_nxt[u] = load_b(brow + (size_t)u * 16 * 512, kc + 32, lh);
    }
#pragma unroll
    for (int u = 0; u < 4; ++u) c[u] = wmma_f16(a_cur, b_cur[u], c[u]);
    a_cur = a_nxt;
#pragma unroll
    for (int u = 0; u < 4; ++u) b_cur[u] = b_nxt[u];
  }
  int rbase = rt * 16 + 8 * lh;
#pragma unroll
  for (int u = 0; u < 4; ++u) {
    int col = ctg * 64 + u * 16 + ll;
    float bv = b1[col];
    float s = 0.f, q = 0.f;
#pragma unroll
    for (int v = 0; v < 8; ++v) {
      float f = fmaxf(c[u][v] + bv, 0.f);
      h1[(size_t)(rbase + v) * 256 + col] = (_Float16)f;
      s += f; q += f * f;
    }
    atomicAdd(&sum1[col], s);
    atomicAdd(&sumsq1[col], q);
  }
}

__global__ __launch_bounds__(256) void bn_finalize_kernel(
    const float* __restrict__ sum, const float* __restrict__ sumsq,
    const float* __restrict__ g, const float* __restrict__ be, float cnt_inv,
    float* __restrict__ scale, float* __restrict__ shift) {
  int c = threadIdx.x;
  float m = sum[c] * cnt_inv;
  float var = fmaxf(sumsq[c] * cnt_inv - m * m, 0.f);
  float sc = g[c] * rsqrtf(var + 1e-5f);
  scale[c] = sc;
  shift[c] = be[c] - m * sc;
}

// -------------- edge GEMM 2: 131072 x 256 x 256 (fused BN, 16x64 tiles,
// software-pipelined)

__global__ __launch_bounds__(128) void edge_gemm2_kernel(
    const _Float16* __restrict__ h1, const float* __restrict__ scale1,
    const float* __restrict__ shift1, const _Float16* __restrict__ W2t,
    const float* __restrict__ b2, _Float16* __restrict__ h2,
    float* __restrict__ sum2, float* __restrict__ sumsq2) {
  int lane = threadIdx.x, lh = lane >> 4, ll = lane & 15;
  int rt = blockIdx.x * 4 + threadIdx.y;
  int ctg = blockIdx.y;
  const _Float16* arow = h1 + (size_t)(rt * 16 + ll) * 256;
  const _Float16* brow = W2t + (size_t)(ctg * 64 + ll) * 256;
  v8f c[4] = {{}, {}, {}, {}};
  v16h a_cur = make_a(load_bn8(arow, scale1, shift1, lh * 8),
                      load_bn8(arow, scale1, shift1, 16 + lh * 8));
  v16h b_cur[4];
#pragma unroll
  for (int u = 0; u < 4; ++u) b_cur[u] = load_b(brow + (size_t)u * 16 * 256, 0, lh);
#pragma unroll
  for (int kc = 0; kc < 256; kc += 32) {
    v16h a_nxt = a_cur;
    v16h b_nxt[4] = {b_cur[0], b_cur[1], b_cur[2], b_cur[3]};
    if (kc + 32 < 256) {
      a_nxt = make_a(load_bn8(arow, scale1, shift1, kc + 32 + lh * 8),
                     load_bn8(arow, scale1, shift1, kc + 48 + lh * 8));
#pragma unroll
      for (int u = 0; u < 4; ++u)
        b_nxt[u] = load_b(brow + (size_t)u * 16 * 256, kc + 32, lh);
    }
#pragma unroll
    for (int u = 0; u < 4; ++u) c[u] = wmma_f16(a_cur, b_cur[u], c[u]);
    a_cur = a_nxt;
#pragma unroll
    for (int u = 0; u < 4; ++u) b_cur[u] = b_nxt[u];
  }
  int rbase = rt * 16 + 8 * lh;
#pragma unroll
  for (int u = 0; u < 4; ++u) {
    int col = ctg * 64 + u * 16 + ll;
    float bv = b2[col];
    float s = 0.f, q = 0.f;
#pragma unroll
    for (int v = 0; v < 8; ++v) {
      float f = fmaxf(c[u][v] + bv, 0.f);
      h2[(size_t)(rbase + v) * 256 + col] = (_Float16)f;
      s += f; q += f * f;
    }
    atomicAdd(&sum2[col], s);
    atomicAdd(&sumsq2[col], q);
  }
}

// ------------------------------- max over K then N (BN2 applied on the fly)

__global__ __launch_bounds__(256) void pool_kernel(
    const _Float16* __restrict__ h2, const float* __restrict__ scale2,
    const float* __restrict__ shift2, float* __restrict__ pooled) {
  int c = threadIdx.x;
  int b = blockIdx.x;
  int n0 = blockIdx.y * 128;
  float sc = scale2[c], sh = shift2[c];
  float m = -3.0e38f;
  for (int n = 0; n < 128; ++n) {
    const _Float16* rows = h2 + (size_t)(b * N_ + n0 + n) * K_ * 256;
    __builtin_prefetch((const void*)(rows + (size_t)K_ * 256), 0, 0);  // next node
#pragma unroll
    for (int k = 0; k < K_; ++k)
      m = fmaxf(m, (float)rows[k * 256 + c] * sc + sh);
  }
  atomicMaxFloat(&pooled[b * 256 + c], m);
}

// ------------------------------- head: 2x (256x256 linear+relu) + l2norm

__global__ __launch_bounds__(256) void head_kernel(
    const float* __restrict__ pooled, const float* __restrict__ W1,
    const float* __restrict__ b1, const float* __restrict__ W2,
    const float* __restrict__ b2, float* __restrict__ out) {
  __shared__ float p[256];
  __shared__ float h[256];
  __shared__ float red[8];
  int c = threadIdx.x, b = blockIdx.x;
  p[c] = pooled[b * 256 + c];
  __syncthreads();
  float acc = b1[c];
  for (int i = 0; i < 256; ++i) acc += p[i] * W1[i * 256 + c];
  h[c] = fmaxf(acc, 0.f);
  __syncthreads();
  acc = b2[c];
  for (int i = 0; i < 256; ++i) acc += h[i] * W2[i * 256 + c];
  float o = fmaxf(acc, 0.f);
  float ss = o * o;
#pragma unroll
  for (int m = 16; m; m >>= 1) ss += __shfl_xor(ss, m, 32);
  if ((c & 31) == 0) red[c >> 5] = ss;
  __syncthreads();
  if (c == 0) { float t = 0.f; for (int i = 0; i < 8; ++i) t += red[i]; red[0] = t; }
  __syncthreads();
  out[b * 256 + c] = o / fmaxf(sqrtf(red[0]), 1e-12f);
}

// ---------------------------------------------------------------- launcher

extern "C" void kernel_launch(void* const* d_in, const int* in_sizes, int n_in,
                              void* d_out, int out_size, void* d_ws, size_t ws_size,
                              hipStream_t stream) {
  (void)in_sizes; (void)n_in; (void)out_size; (void)ws_size;
  const int*   cls       = (const int*)d_in[0];
  const float* colors    = (const float*)d_in[1];
  const float* positions = (const float*)d_in[2];
  const float* table     = (const float*)d_in[3];
  const float* posW1 = (const float*)d_in[4],  *posb1 = (const float*)d_in[5];
  const float* posW2 = (const float*)d_in[6],  *posb2 = (const float*)d_in[7];
  const float* colW1 = (const float*)d_in[8],  *colb1 = (const float*)d_in[9];
  const float* colW2 = (const float*)d_in[10], *colb2 = (const float*)d_in[11];
  const float* mergeW = (const float*)d_in[12], *mergeb = (const float*)d_in[13];
  const float* gW1 = (const float*)d_in[14], *gb1 = (const float*)d_in[15];
  const float* gg1 = (const float*)d_in[16], *gbe1 = (const float*)d_in[17];
  const float* gW2 = (const float*)d_in[18], *gb2 = (const float*)d_in[19];
  const float* gg2 = (const float*)d_in[20], *gbe2 = (const float*)d_in[21];
  const float* linW1 = (const float*)d_in[22], *linb1 = (const float*)d_in[23];
  const float* linW2 = (const float*)d_in[24], *linb2 = (const float*)d_in[25];
  float* out = (float*)d_out;

  char* base = (char*)d_ws;
  size_t off = 0;
  auto take = [&](size_t bytes) -> char* {
    char* p = base + off;
    off = (off + bytes + 255) & ~(size_t)255;
    return p;
  };
  _Float16* mergeWt = (_Float16*)take((size_t)256 * 768 * 2);
  _Float16* gW1t    = (_Float16*)take((size_t)256 * 512 * 2);
  _Float16* gW2t    = (_Float16*)take((size_t)256 * 256 * 2);
  _Float16* feat    = (_Float16*)take((size_t)P_ * 768 * 2);
  _Float16* xh      = (_Float16*)take((size_t)P_ * 256 * 2);
  float*    sq      = (float*)take((size_t)P_ * 4);
  int*      knn     = (int*)take((size_t)P_ * K_ * 4);
  _Float16* h1      = (_Float16*)take((size_t)E_ * 256 * 2);
  _Float16* h2      = (_Float16*)take((size_t)E_ * 256 * 2);
  float* sum1   = (float*)take(256 * 4);
  float* sumsq1 = (float*)take(256 * 4);
  float* scale1 = (float*)take(256 * 4);
  float* shift1 = (float*)take(256 * 4);
  float* sum2   = (float*)take(256 * 4);
  float* sumsq2 = (float*)take(256 * 4);
  float* scale2 = (float*)take(256 * 4);
  float* shift2 = (float*)take(256 * 4);
  float* pooled = (float*)take((size_t)B_ * 256 * 4);

  transpose_f16_kernel<<<(768 * 256 + 255) / 256, 256, 0, stream>>>(mergeW, mergeWt, 768, 256);
  transpose_f16_kernel<<<(512 * 256 + 255) / 256, 256, 0, stream>>>(gW1, gW1t, 512, 256);
  transpose_f16_kernel<<<(256 * 256 + 255) / 256, 256, 0, stream>>>(gW2, gW2t, 256, 256);
  init_kernel<<<(B_ * D_ + 255) / 256, 256, 0, stream>>>(pooled, sum1, sumsq1, sum2, sumsq2);

  embed_kernel<<<P_ / 8, 256, 0, stream>>>(cls, colors, positions, table,
      posW1, posb1, posW2, posb2, colW1, colb1, colW2, colb2, feat);
  merge_gemm_kernel<<<dim3(P_ / 64, 4), dim3(32, 4), 0, stream>>>(feat, mergeWt, mergeb, xh);
  sq_kernel<<<P_ / 8, 256, 0, stream>>>(xh, sq);
  knn_kernel<<<B_ * (N_ / 16), 32, 0, stream>>>(xh, sq, knn);
  edge_gemm1_kernel<<<dim3(E_ / 64, 4), dim3(32, 4), 0, stream>>>(xh, knn, gW1t, gb1,
                                                                  h1, sum1, sumsq1);
  bn_finalize_kernel<<<1, 256, 0, stream>>>(sum1, sumsq1, gg1, gbe1,
                                            1.0f / (float)E_, scale1, shift1);
  edge_gemm2_kernel<<<dim3(E_ / 64, 4), dim3(32, 4), 0, stream>>>(h1, scale1, shift1,
                                                                  gW2t, gb2, h2, sum2, sumsq2);
  bn_finalize_kernel<<<1, 256, 0, stream>>>(sum2, sumsq2, gg2, gbe2,
                                            1.0f / (float)E_, scale2, shift2);
  pool_kernel<<<dim3(B_, N_ / 128), 256, 0, stream>>>(h2, scale2, shift2, pooled);
  head_kernel<<<B_, 256, 0, stream>>>(pooled, linW1, linb1, linW2, linb2, out);
}
